// SimpleGNNEncoder_35287451304146
// MI455X (gfx1250) — matrix operand order repaired and checked by reading, not compiled, not run on hardware
//
#include <hip/hip_runtime.h>
#include <hip/hip_bf16.h>

// ---------------------------------------------------------------------------
// SimpleGNNEncoder on MI455X (gfx1250), wave32 + WMMA bf16 + async LDS copy.
//
// Shapes: N=50000 nodes, E=600000 edges, H=128, 3 layers.
//   h = relu(x @ Wp + bp)
//   per layer: m = relu(cat(h[dst],h[src]) @ msg_W + msg_b)   [E,128]
//              agg = scatter_add(m, dst)                       [N,128]
//              h = relu(agg @ lin_W + lin_b)
//   out = mean(h, axis=0)                                      [1,128]
//
// GEMMs via v_wmma_f32_16x16x32_bf16. The edge GEMM (dominant, ~118 GFLOP
// total) stages its gathered 16x256 A-tile in LDS once per block via
// global_load_async_to_lds_b128 (ASYNCcnt) and reads fragments with
// ds_load_b128, eliminating the 8x inter-wave gather redundancy.
// ---------------------------------------------------------------------------

typedef __attribute__((ext_vector_type(16))) __bf16 v16bf;
typedef __attribute__((ext_vector_type(8)))  __bf16 v8bf;
typedef __attribute__((ext_vector_type(8)))  float  v8f;
typedef __attribute__((ext_vector_type(4)))  int    v4i;

#define NN   50000
#define NE   600000
#define HD   128

// Padded A-tile row length in halves: 256 data + 8 pad -> 528B row stride,
// 132-dword stride = 4-bank rotation per row -> conflict-free ds_load_b128.
#define ROWP 264

#if __has_builtin(__builtin_amdgcn_global_load_async_to_lds_b128)
#define HAVE_ASYNC_LDS 1
#else
#define HAVE_ASYNC_LDS 0
#endif

static __device__ __forceinline__ void async_copy_b128(const void* g, void* l) {
#if HAVE_ASYNC_LDS
    // Builtin signature (from hipcc diagnostic): param 0 is a global
    // (__device__/addrspace(1)) pointer to 'int __vector(4)'; LDS side is the
    // addrspace(3) counterpart.
    __builtin_amdgcn_global_load_async_to_lds_b128(
        (__attribute__((address_space(1))) v4i*)g,
        (__attribute__((address_space(3))) v4i*)l,
        /*offset=*/0, /*cpol=*/0);
#else
    *(uint4*)l = *(const uint4*)g;   // fallback: VGPR bounce
#endif
}

static __device__ __forceinline__ void wait_async_and_barrier() {
#if HAVE_ASYNC_LDS
#if __has_builtin(__builtin_amdgcn_s_wait_asynccnt)
    __builtin_amdgcn_s_wait_asynccnt(0);
#else
    asm volatile("s_wait_asynccnt 0x0" ::: "memory");
#endif
#endif
    __syncthreads();
}

static __device__ __forceinline__ v8f wmma_bf16(v16bf a, v16bf b, v8f c) {
    return __builtin_amdgcn_wmma_f32_16x16x32_bf16(
        /*neg_a=*/false, a, /*neg_b=*/false, b,
        /*c_mod=*/(short)0, c, /*reuse_a=*/false, /*reuse_b=*/false);
}

// ---------------------------------------------------------------------------
// Pre-pack an f32 weight matrix W[Ktot,128] (row-major, N contiguous) into
// B-fragment layout for v_wmma_*_bf16:
//   fragment f = kt*8 + nt   (kt = K/32 tile, nt = N/16 tile)
//   lane l: N = nt*16 + (l%16), K = kt*32 + (l/16)*16 + e,  e = 0..15
//   packed[( f*32 + l )*16 + e]
// ---------------------------------------------------------------------------
__global__ __launch_bounds__(256) void prepack_w(const float* __restrict__ W,
                                                 __bf16* __restrict__ out,
                                                 int ktiles) {
    int t    = blockIdx.x * 256 + threadIdx.x;
    int nf   = ktiles * 8;
    int f    = t >> 5;
    int lane = t & 31;
    if (f >= nf) return;
    int kt = f >> 3, nt = f & 7;
    int N     = nt * 16 + (lane & 15);
    int Kbase = kt * 32 + (lane >> 4) * 16;
    __bf16* dst = out + ((size_t)f * 32 + lane) * 16;
    v16bf v;
    #pragma unroll
    for (int e = 0; e < 16; ++e)
        v[e] = (__bf16)W[(size_t)(Kbase + e) * HD + N];
    *(v16bf*)dst = v;
}

// ---------------------------------------------------------------------------
// Node GEMM: out[M,128](bf16) = relu( A[M,128](f32) @ Wpk + bias )
// grid = M/16 blocks, 256 threads = 8 waves; wave w -> N-tile w.
// ---------------------------------------------------------------------------
__global__ __launch_bounds__(256) void node_gemm(const float* __restrict__ A,
                                                 const __bf16* __restrict__ Wpk,
                                                 const float* __restrict__ bias,
                                                 __bf16* __restrict__ out,
                                                 int M) {
    int tile = blockIdx.x;
    int t    = threadIdx.x;
    int nt   = t >> 5;          // wave id = N tile
    int lane = t & 31;
    int g    = lane >> 4;       // K-half selector
    int r    = lane & 15;       // row within tile (A) / column (C)

    float bv = bias[nt * 16 + r];
    v8f c;
    #pragma unroll
    for (int i = 0; i < 8; ++i) c[i] = bv;   // accumulator pre-loaded with bias

    const float* Arow = A + (size_t)(tile * 16 + r) * HD;
    const v16bf* Bf   = (const v16bf*)Wpk;

    #pragma unroll
    for (int kt = 0; kt < 4; ++kt) {         // K = 128 = 4 * 32
        int K1 = kt * 32 + g * 16;
        float4 f0 = *(const float4*)(Arow + K1);
        float4 f1 = *(const float4*)(Arow + K1 + 4);
        float4 f2 = *(const float4*)(Arow + K1 + 8);
        float4 f3 = *(const float4*)(Arow + K1 + 12);
        v16bf a;
        a[0]=(__bf16)f0.x; a[1]=(__bf16)f0.y; a[2]=(__bf16)f0.z; a[3]=(__bf16)f0.w;
        a[4]=(__bf16)f1.x; a[5]=(__bf16)f1.y; a[6]=(__bf16)f1.z; a[7]=(__bf16)f1.w;
        a[8]=(__bf16)f2.x; a[9]=(__bf16)f2.y; a[10]=(__bf16)f2.z; a[11]=(__bf16)f2.w;
        a[12]=(__bf16)f3.x; a[13]=(__bf16)f3.y; a[14]=(__bf16)f3.z; a[15]=(__bf16)f3.w;
        v16bf b = Bf[kt * 8 + nt];
        c = wmma_bf16(a, b, c);
    }

    // C fragment: element i -> M = tile*16 + g*8 + i, N = nt*16 + r
    #pragma unroll
    for (int i = 0; i < 8; ++i) {
        float v = fmaxf(c[i], 0.0f);
        int m = tile * 16 + g * 8 + i;
        out[(size_t)m * HD + nt * 16 + r] = (__bf16)v;
    }
}

// ---------------------------------------------------------------------------
// Edge message GEMM + scatter-add:
//   m[e,:] = relu( cat(h[dst[e]], h[src[e]]) @ msgW + msg_b )
//   agg[dst[e],:] += m[e,:]          (global_atomic_add_f32, agg L2-resident)
//
// Block = 16 edges. The 16x256 bf16 A-tile (rows = h[dst[e]] ‖ h[src[e]])
// is gathered into LDS exactly once via async b128 copies (512 chunks,
// 2 per thread), then all 8 waves read fragments from LDS.
// ---------------------------------------------------------------------------
__global__ __launch_bounds__(256) void edge_gemm_scatter(
        const __bf16* __restrict__ h,
        const long long* __restrict__ ei,     // [2,E]: row0 = src, row1 = dst
        const __bf16* __restrict__ Wpk,       // packed msg_W (8 K-tiles)
        const float* __restrict__ bias,
        float* __restrict__ agg) {
    __shared__ __align__(16) __bf16 sA[16 * ROWP];  // 16 rows x 264 halves
    __shared__ long long s_dst[16];

    int tile = blockIdx.x;
    int t    = threadIdx.x;

    if (t < 16)
        s_dst[t] = ei[(size_t)NE + tile * 16 + t];  // aggregation index

    // Stage A-tile: chunk c = (row, part); part 0..15 -> h[dst] K=0..127,
    // part 16..31 -> h[src] K=128..255. Each chunk = 8 bf16 = 16 bytes.
    #pragma unroll
    for (int c = t; c < 512; c += 256) {
        int row  = c >> 5;
        int part = c & 31;
        long long node = (part < 16) ? ei[(size_t)NE + tile * 16 + row]   // dst
                                     : ei[(size_t)tile * 16 + row];       // src
        const __bf16* g = h + (size_t)node * HD + (size_t)(part & 15) * 8;
        async_copy_b128(g, &sA[row * ROWP + part * 8]);
    }
    wait_async_and_barrier();

    int nt   = t >> 5;
    int lane = t & 31;
    int g    = lane >> 4;
    int r    = lane & 15;

    float bv = bias[nt * 16 + r];
    v8f c;
    #pragma unroll
    for (int i = 0; i < 8; ++i) c[i] = bv;

    const v16bf* Bf = (const v16bf*)Wpk;

    #pragma unroll
    for (int kt = 0; kt < 8; ++kt) {          // K = 256 = 8 * 32
        int p1 = kt * 4 + g * 2;              // chunk index of K = kt*32 + g*16
        v8bf ca = *(const v8bf*)&sA[r * ROWP + p1 * 8];        // ds_load_b128
        v8bf cb = *(const v8bf*)&sA[r * ROWP + (p1 + 1) * 8];  // K + 8
        v16bf a;
        #pragma unroll
        for (int i = 0; i < 8; ++i) { a[i] = ca[i]; a[8 + i] = cb[i]; }
        v16bf b = Bf[kt * 8 + nt];
        c = wmma_bf16(a, b, c);
    }

    // relu then scatter-add each row of the 16x16 tile into agg[dst[row]]
    #pragma unroll
    for (int i = 0; i < 8; ++i) {
        float v = fmaxf(c[i], 0.0f);
        int row = g * 8 + i;                  // edge within tile
        long long node = s_dst[row];
        atomicAdd(&agg[(size_t)node * HD + nt * 16 + r], v);
    }
}

// ---------------------------------------------------------------------------
// Zero fill (for agg before the atomic scatter) and final column mean.
// ---------------------------------------------------------------------------
__global__ __launch_bounds__(256) void zero_f32(float4* __restrict__ p, int n4) {
    int i = blockIdx.x * 256 + threadIdx.x;
    if (i < n4) p[i] = make_float4(0.f, 0.f, 0.f, 0.f);
}

__global__ __launch_bounds__(256) void col_mean(const __bf16* __restrict__ h,
                                                float* __restrict__ out, int M) {
    __shared__ float s[256];
    int col = blockIdx.x;
    int t   = threadIdx.x;
    float acc = 0.f;
    for (int n = t; n < M; n += 256)
        acc += (float)h[(size_t)n * HD + col];
    s[t] = acc;
    __syncthreads();
    for (int off = 128; off > 0; off >>= 1) {
        if (t < off) s[t] += s[t + off];
        __syncthreads();
    }
    if (t == 0) out[col] = s[0] / (float)M;
}

// ---------------------------------------------------------------------------
// Orchestration.
// Inputs (setup_inputs order):
//   0: x        [N,128]      f32
//   1: edge_idx [2,E]        i64
//   2: Wp       [128,128]    f32
//   3: bp       [128]        f32
//   4: msg_W    [3,256,128]  f32
//   5: msg_b    [3,128]      f32
//   6: lin_W    [3,128,128]  f32
//   7: lin_b    [3,128]      f32
// Output: [1,128] f32.
// Workspace: h(bf16 12.8MB) | agg(f32 25.6MB) | packed weights (~0.33MB)
// ---------------------------------------------------------------------------
extern "C" void kernel_launch(void* const* d_in, const int* in_sizes, int n_in,
                              void* d_out, int out_size, void* d_ws, size_t ws_size,
                              hipStream_t stream) {
    const float*     x    = (const float*)d_in[0];
    const long long* ei   = (const long long*)d_in[1];
    const float*     Wp   = (const float*)d_in[2];
    const float*     bp   = (const float*)d_in[3];
    const float*     msgW = (const float*)d_in[4];
    const float*     msgb = (const float*)d_in[5];
    const float*     linW = (const float*)d_in[6];
    const float*     linb = (const float*)d_in[7];
    float*           out  = (float*)d_out;

    char* ws = (char*)d_ws;
    __bf16* h    = (__bf16*)ws;                                    // N*H bf16
    float*  agg  = (float*)(ws + (size_t)NN * HD * 2);             // N*H f32
    __bf16* WpP  = (__bf16*)(ws + (size_t)NN * HD * 2 + (size_t)NN * HD * 4);
    __bf16* msgP = WpP  + (size_t)HD * HD;                         // 3 * 256*128
    __bf16* linP = msgP + (size_t)3 * 2 * HD * HD;                 // 3 * 128*128

    // --- Pre-pack all weights into WMMA B-fragment layout (bf16) ---
    prepack_w<<<4, 256, 0, stream>>>(Wp, WpP, /*ktiles=*/4);
    for (int l = 0; l < 3; ++l) {
        prepack_w<<<8, 256, 0, stream>>>(msgW + (size_t)l * 2 * HD * HD,
                                         msgP + (size_t)l * 2 * HD * HD, /*ktiles=*/8);
        prepack_w<<<4, 256, 0, stream>>>(linW + (size_t)l * HD * HD,
                                         linP + (size_t)l * HD * HD, /*ktiles=*/4);
    }

    // --- Input projection: h = relu(x @ Wp + bp) ---
    node_gemm<<<NN / 16, 256, 0, stream>>>(x, WpP, bp, h, NN);

    // --- 3 message-passing layers ---
    for (int l = 0; l < 3; ++l) {
        zero_f32<<<(NN * HD / 4 + 255) / 256, 256, 0, stream>>>((float4*)agg,
                                                                NN * HD / 4);
        edge_gemm_scatter<<<NE / 16, 256, 0, stream>>>(
            h, ei, msgP + (size_t)l * 2 * HD * HD, msgb + (size_t)l * HD, agg);
        node_gemm<<<NN / 16, 256, 0, stream>>>(
            agg, linP + (size_t)l * HD * HD, linb + (size_t)l * HD, h, NN);
    }

    // --- Global mean over nodes -> [1,128] ---
    col_mean<<<HD, 256, 0, stream>>>(h, out, NN);
}
